// SlidingWindowAttention_49031346651481
// MI455X (gfx1250) — matrix operand order repaired and checked by reading, not compiled, not run on hardware
//
#include <hip/hip_runtime.h>

// ---------------------------------------------------------------------------
// Sliding-window attention block for gfx1250 (MI455X), bf16 WMMA pipeline.
//   1) convert x -> bf16, W_qkv/W_proj -> bf16 transposed [N][K] (tiled)
//   2) QKV GEMM (wmma bf16, f32 acc, async-to-LDS double buffer)
//        -> Q[h][t][d], K[h][t][d], V^T[h][d][t]
//   3) flash-style windowed attention (wmma QK^T + online softmax + wmma PV)
//   4) proj GEMM (wmma, async-to-LDS) -> fp32 output with bias
// ---------------------------------------------------------------------------

#define T_SEQ 4096
#define CDIM  1024
#define NH    16
#define HD    64
#define N3    3072
#define WIN   256

typedef __attribute__((ext_vector_type(16))) __bf16 v16bf;
typedef __attribute__((ext_vector_type(4)))  __bf16 v4bf;
typedef __attribute__((ext_vector_type(8)))  float  v8f;
typedef __attribute__((ext_vector_type(4)))  int    v4i;

union Frag {
  v16bf v;
  __bf16 h[16];
  uint4 u[2];
};

static __device__ __forceinline__ v8f wmma_bf16(v16bf a, v16bf b, v8f c) {
  // D = A(16x32) * B(32x16) + C, f32 accumulate
  return __builtin_amdgcn_wmma_f32_16x16x32_bf16(false, a, false, b,
                                                 (short)0, c, false, false);
}

static __device__ __forceinline__ v8f v8zero() {
  v8f z;
  for (int i = 0; i < 8; i++) z[i] = 0.0f;
  return z;
}

// ---------------- async global->LDS copy (32 bytes per call) ---------------

#if defined(__gfx1250__) && __has_builtin(__builtin_amdgcn_global_load_async_to_lds_b128)
#define USE_ASYNC 1
#else
#define USE_ASYNC 0
#endif

#define GAS __attribute__((address_space(1)))
#define LAS __attribute__((address_space(3)))

static __device__ __forceinline__ void cp32(const __bf16* g, __bf16* l) {
#if USE_ASYNC
  __builtin_amdgcn_global_load_async_to_lds_b128(
      (GAS v4i*)(GAS void*)g, (LAS v4i*)(LAS void*)l, 0, 0);
  __builtin_amdgcn_global_load_async_to_lds_b128(
      (GAS v4i*)(GAS void*)(g + 8), (LAS v4i*)(LAS void*)(l + 8), 0, 0);
#else
  *(uint4*)l = *(const uint4*)g;
  *(uint4*)(l + 8) = *(const uint4*)(g + 8);
#endif
}

static __device__ __forceinline__ void wait_async() {
#if USE_ASYNC
#if __has_builtin(__builtin_amdgcn_s_wait_asynccnt)
  __builtin_amdgcn_s_wait_asynccnt(0);
#else
  asm volatile("s_wait_asynccnt 0x0" ::: "memory");
#endif
#endif
}

// -------------------------- conversion kernels -----------------------------

__global__ void cvt_bf16(const float4* __restrict__ in, v4bf* __restrict__ out,
                         int n4) {
  for (int i = blockIdx.x * blockDim.x + threadIdx.x; i < n4;
       i += gridDim.x * blockDim.x) {
    float4 f = in[i];
    v4bf o;
    o[0] = (__bf16)f.x; o[1] = (__bf16)f.y;
    o[2] = (__bf16)f.z; o[3] = (__bf16)f.w;
    out[i] = o;
  }
}

// in: fp32 [R][C] -> out: bf16 [C][R], 32x32 LDS tiles, coalesced both sides
__global__ __launch_bounds__(256) void transpose_bf16(
    const float* __restrict__ in, __bf16* __restrict__ out, int R, int C) {
  __shared__ float tile[32][33];
  const int tx = threadIdx.x & 31, ty = threadIdx.x >> 5;  // ty: 0..7
  const int c0 = blockIdx.x * 32, r0 = blockIdx.y * 32;
  for (int j = 0; j < 32; j += 8)
    tile[ty + j][tx] = in[(size_t)(r0 + ty + j) * C + c0 + tx];
  __syncthreads();
  for (int j = 0; j < 32; j += 8)
    out[(size_t)(c0 + ty + j) * R + r0 + tx] = (__bf16)tile[tx][ty + j];
}

// ------------------------------ QKV GEMM -----------------------------------
// A: x_bf16 [T_SEQ][CDIM], Bt: Wqkv^T bf16 [N3][CDIM], bias fp32 [N3]
// Block tile 128x128, 8 waves of 32x64, K-step 32, double-buffered LDS.

__global__ __launch_bounds__(256) void gemm_qkv(
    const __bf16* __restrict__ A, const __bf16* __restrict__ Bt,
    const float* __restrict__ bias,
    __bf16* __restrict__ Qb, __bf16* __restrict__ Kb, __bf16* __restrict__ Vt) {
  __shared__ alignas(16) __bf16 sA[2][128 * 32];
  __shared__ alignas(16) __bf16 sB[2][128 * 32];
  const int tid = threadIdx.x;
  const int wave = tid >> 5, lane = tid & 31;
  const int lg = lane >> 4, ln = lane & 15;
  const int wm = wave >> 1, wn = wave & 1;
  const int m0 = blockIdx.y * 128, n0 = blockIdx.x * 128;
  const int lrow = tid >> 1, lseg = (tid & 1) * 16;
  const int loff = lrow * 32 + lseg;

  v8f acc[2][4];
  for (int a = 0; a < 2; a++)
    for (int b = 0; b < 4; b++) acc[a][b] = v8zero();

  const __bf16* gA = A + (size_t)(m0 + lrow) * CDIM + lseg;
  const __bf16* gB = Bt + (size_t)(n0 + lrow) * CDIM + lseg;

  // preload first K-tile into buffer 0
  cp32(gA, &sA[0][loff]);
  cp32(gB, &sB[0][loff]);

  int s = 0;
  for (int k0 = 0; k0 < CDIM; k0 += 32, s ^= 1) {
    wait_async();
    __syncthreads();  // publishes buf[s]; retires all reads of buf[s^1]
    if (k0 + 32 < CDIM) {
      cp32(gA + k0 + 32, &sA[s ^ 1][loff]);
      cp32(gB + k0 + 32, &sB[s ^ 1][loff]);
    }
    Frag af[2], bfr[4];
    for (int mt = 0; mt < 2; mt++) {
      const __bf16* p = &sA[s][(wm * 32 + mt * 16 + ln) * 32 + lg * 8];
      af[mt].u[0] = *(const uint4*)p;
      af[mt].u[1] = *(const uint4*)(p + 16);
    }
    for (int nt = 0; nt < 4; nt++) {
      const __bf16* p = &sB[s][(wn * 64 + nt * 16 + ln) * 32 + lg * 8];
      bfr[nt].u[0] = *(const uint4*)p;
      bfr[nt].u[1] = *(const uint4*)(p + 16);
    }
    for (int mt = 0; mt < 2; mt++)
      for (int nt = 0; nt < 4; nt++)
        acc[mt][nt] = wmma_bf16(af[mt].v, bfr[nt].v, acc[mt][nt]);
  }

  // epilogue: bias add + scatter into Q [h][t][d], K [h][t][d], V^T [h][d][t]
  for (int mt = 0; mt < 2; mt++)
    for (int nt = 0; nt < 4; nt++) {
      int gn = n0 + wn * 64 + nt * 16 + ln;
      float bv = bias[gn];
      int sec = gn >> 10;  // 0=q 1=k 2=v
      int hh = (gn >> 6) & (NH - 1);
      int d = gn & (HD - 1);
      for (int i = 0; i < 8; i++) {
        int gm = m0 + wm * 32 + mt * 16 + lg * 8 + i;
        __bf16 val = (__bf16)(acc[mt][nt][i] + bv);
        if (sec == 0)      Qb[((size_t)hh * T_SEQ + gm) * HD + d] = val;
        else if (sec == 1) Kb[((size_t)hh * T_SEQ + gm) * HD + d] = val;
        else               Vt[((size_t)hh * HD + d) * T_SEQ + gm] = val;
      }
    }
}

// ----------------------------- attention -----------------------------------
// One wave = 16 queries of one head, flash-style over 32-key tiles.
// grid: (T_SEQ/128, NH), block 256 (8 waves, no cross-wave sync needed).

__global__ __launch_bounds__(256) void attn_kernel(
    const __bf16* __restrict__ Qb, const __bf16* __restrict__ Kb,
    const __bf16* __restrict__ Vt, __bf16* __restrict__ Y) {
  __shared__ alignas(16) __bf16 sP[8 * 16 * 32];  // per-wave 16x32 P tile
  const int tid = threadIdx.x;
  const int wave = tid >> 5, lane = tid & 31;
  const int lg = lane >> 4, ln = lane & 15;
  const int head = blockIdx.y;
  const int q0 = (blockIdx.x * 8 + wave) * 16;
  __bf16* myP = sP + wave * 512;

  // Q A-frags (two 16x32 chunks covering d=0..63)
  Frag qa[2];
  {
    const __bf16* qp = Qb + ((size_t)head * T_SEQ + (q0 + ln)) * HD;
    for (int c = 0; c < 2; c++) {
      const __bf16* p = qp + c * 32 + lg * 8;
      qa[c].u[0] = *(const uint4*)p;
      qa[c].u[1] = *(const uint4*)(p + 16);
    }
  }

  float rm[8], rl[8];
  v8f o[4];
  for (int i = 0; i < 8; i++) { rm[i] = -1e30f; rl[i] = 0.0f; }
  for (int c = 0; c < 4; c++) o[c] = v8zero();

  int kstart = q0 - (WIN - 1);
  if (kstart < 0) kstart = 0;
  kstart &= ~31;

  for (int k0 = kstart; k0 <= q0 + 15; k0 += 32) {
    // prefetch next key tile (K rows + V^T rows) into caches
    if (k0 + 32 <= q0 + 15) {
      int nk = k0 + 32 + lane;
      if (nk > T_SEQ - 1) nk = T_SEQ - 1;
      __builtin_prefetch(Kb + ((size_t)head * T_SEQ + nk) * HD, 0, 1);
      __builtin_prefetch(Vt + ((size_t)head * HD + lane) * T_SEQ + (k0 + 32), 0, 1);
      __builtin_prefetch(Vt + ((size_t)head * HD + 32 + lane) * T_SEQ + (k0 + 32), 0, 1);
    }

    // ---- S = Q * K^T (two 16-key groups) ----
    Frag kf[2][2];
    for (int g = 0; g < 2; g++) {
      int key = k0 + g * 16 + ln;
      if (key > T_SEQ - 1) key = T_SEQ - 1;  // clamped; masked below
      const __bf16* kp = Kb + ((size_t)head * T_SEQ + key) * HD + lg * 8;
      for (int c = 0; c < 2; c++) {
        const __bf16* p = kp + c * 32;
        kf[g][c].u[0] = *(const uint4*)p;
        kf[g][c].u[1] = *(const uint4*)(p + 16);
      }
    }
    float sv[2][8];
    for (int g = 0; g < 2; g++) {
      v8f z = v8zero();
      z = wmma_bf16(qa[0].v, kf[g][0].v, z);
      z = wmma_bf16(qa[1].v, kf[g][1].v, z);
      for (int i = 0; i < 8; i++) {
        int qi = q0 + lg * 8 + i;
        int ki = k0 + g * 16 + ln;
        bool ok = (qi >= ki) && (qi - ki < WIN);
        sv[g][i] = ok ? z[i] * 0.125f : -1e30f;
      }
    }

    // ---- online softmax (rows live in half-wave lanes) ----
    float alpha[8];
    for (int i = 0; i < 8; i++) {
      float mx = fmaxf(sv[0][i], sv[1][i]);
      for (int m = 1; m < 16; m <<= 1) mx = fmaxf(mx, __shfl_xor(mx, m, 16));
      float nm = fmaxf(rm[i], mx);
      float a = __expf(rm[i] - nm);
      float p0 = __expf(sv[0][i] - nm);
      float p1 = __expf(sv[1][i] - nm);
      float ps = p0 + p1;
      for (int m = 1; m < 16; m <<= 1) ps += __shfl_xor(ps, m, 16);
      rl[i] = rl[i] * a + ps;
      rm[i] = nm;
      alpha[i] = a;
      // C-layout -> LDS (bf16), to be re-read in A-layout
      myP[(lg * 8 + i) * 32 + ln]      = (__bf16)p0;
      myP[(lg * 8 + i) * 32 + 16 + ln] = (__bf16)p1;
    }
    for (int c = 0; c < 4; c++)
      for (int i = 0; i < 8; i++) o[c][i] *= alpha[i];

    asm volatile("s_wait_dscnt 0x0" ::: "memory");  // wave-private LDS ordering

    Frag pa;
    {
      const __bf16* p = myP + ln * 32 + lg * 8;
      pa.u[0] = *(const uint4*)p;
      pa.u[1] = *(const uint4*)(p + 16);
    }

    // ---- O += P * V  (V stored transposed [h][d][t] -> contiguous B-frags) ----
    int kbv = k0 + lg * 8;
    if (kbv > T_SEQ - 24) kbv = T_SEQ - 24;  // clamp tail reads (masked keys)
    for (int c = 0; c < 4; c++) {
      Frag vf;
      const __bf16* vp = Vt + ((size_t)head * HD + c * 16 + ln) * T_SEQ + kbv;
      vf.u[0] = *(const uint4*)vp;
      vf.u[1] = *(const uint4*)(vp + 16);
      o[c] = wmma_bf16(pa.v, vf.v, o[c]);
    }
  }

  // write y [t][c] bf16 for the projection GEMM
  for (int c = 0; c < 4; c++)
    for (int i = 0; i < 8; i++) {
      int qi = q0 + lg * 8 + i;
      float val = o[c][i] / rl[i];
      Y[(size_t)qi * CDIM + head * HD + c * 16 + ln] = (__bf16)val;
    }
}

// ------------------------------ proj GEMM ----------------------------------
// A: y bf16 [T_SEQ][CDIM], Bt: Wproj^T bf16 [CDIM][CDIM], out fp32 + bias.

__global__ __launch_bounds__(256) void gemm_proj(
    const __bf16* __restrict__ A, const __bf16* __restrict__ Bt,
    const float* __restrict__ bias, float* __restrict__ out) {
  __shared__ alignas(16) __bf16 sA[2][128 * 32];
  __shared__ alignas(16) __bf16 sB[2][128 * 32];
  const int tid = threadIdx.x;
  const int wave = tid >> 5, lane = tid & 31;
  const int lg = lane >> 4, ln = lane & 15;
  const int wm = wave >> 1, wn = wave & 1;
  const int m0 = blockIdx.y * 128, n0 = blockIdx.x * 128;
  const int lrow = tid >> 1, lseg = (tid & 1) * 16;
  const int loff = lrow * 32 + lseg;

  v8f acc[2][4];
  for (int a = 0; a < 2; a++)
    for (int b = 0; b < 4; b++) acc[a][b] = v8zero();

  const __bf16* gA = A + (size_t)(m0 + lrow) * CDIM + lseg;
  const __bf16* gB = Bt + (size_t)(n0 + lrow) * CDIM + lseg;

  cp32(gA, &sA[0][loff]);
  cp32(gB, &sB[0][loff]);

  int s = 0;
  for (int k0 = 0; k0 < CDIM; k0 += 32, s ^= 1) {
    wait_async();
    __syncthreads();
    if (k0 + 32 < CDIM) {
      cp32(gA + k0 + 32, &sA[s ^ 1][loff]);
      cp32(gB + k0 + 32, &sB[s ^ 1][loff]);
    }
    Frag af[2], bfr[4];
    for (int mt = 0; mt < 2; mt++) {
      const __bf16* p = &sA[s][(wm * 32 + mt * 16 + ln) * 32 + lg * 8];
      af[mt].u[0] = *(const uint4*)p;
      af[mt].u[1] = *(const uint4*)(p + 16);
    }
    for (int nt = 0; nt < 4; nt++) {
      const __bf16* p = &sB[s][(wn * 64 + nt * 16 + ln) * 32 + lg * 8];
      bfr[nt].u[0] = *(const uint4*)p;
      bfr[nt].u[1] = *(const uint4*)(p + 16);
    }
    for (int mt = 0; mt < 2; mt++)
      for (int nt = 0; nt < 4; nt++)
        acc[mt][nt] = wmma_bf16(af[mt].v, bfr[nt].v, acc[mt][nt]);
  }

  for (int mt = 0; mt < 2; mt++)
    for (int nt = 0; nt < 4; nt++) {
      int gn = n0 + wn * 64 + nt * 16 + ln;
      float bv = bias[gn];
      for (int i = 0; i < 8; i++) {
        int gm = m0 + wm * 32 + mt * 16 + lg * 8 + i;
        out[(size_t)gm * CDIM + gn] = acc[mt][nt][i] + bv;
      }
    }
}

// ------------------------------ launcher -----------------------------------

extern "C" void kernel_launch(void* const* d_in, const int* in_sizes, int n_in,
                              void* d_out, int out_size, void* d_ws, size_t ws_size,
                              hipStream_t stream) {
  const float* x     = (const float*)d_in[0];  // [1,4096,1024]
  const float* Wqkv  = (const float*)d_in[1];  // [1024,3072]
  const float* bqkv  = (const float*)d_in[2];  // [3072]
  const float* Wproj = (const float*)d_in[3];  // [1024,1024]
  const float* bproj = (const float*)d_in[4];  // [1024]
  float* out = (float*)d_out;

  char* ws = (char*)d_ws;
  size_t off = 0;
  auto alloc = [&](size_t bytes) {
    char* p = ws + off;
    off = (off + bytes + 255) & ~(size_t)255;
    return p;
  };
  __bf16* x_bf    = (__bf16*)alloc((size_t)T_SEQ * CDIM * 2);
  __bf16* Wqkv_t  = (__bf16*)alloc((size_t)N3 * CDIM * 2);
  __bf16* Wproj_t = (__bf16*)alloc((size_t)CDIM * CDIM * 2);
  __bf16* Qb      = (__bf16*)alloc((size_t)NH * T_SEQ * HD * 2);
  __bf16* Kb      = (__bf16*)alloc((size_t)NH * T_SEQ * HD * 2);
  __bf16* Vt      = (__bf16*)alloc((size_t)NH * HD * T_SEQ * 2);
  __bf16* Yb      = (__bf16*)alloc((size_t)T_SEQ * CDIM * 2);
  (void)ws_size; (void)in_sizes; (void)n_in; (void)out_size;

  cvt_bf16<<<1024, 256, 0, stream>>>((const float4*)x, (v4bf*)x_bf,
                                     (T_SEQ * CDIM) / 4);
  transpose_bf16<<<dim3(N3 / 32, CDIM / 32), 256, 0, stream>>>(Wqkv, Wqkv_t,
                                                               CDIM, N3);
  transpose_bf16<<<dim3(CDIM / 32, CDIM / 32), 256, 0, stream>>>(Wproj, Wproj_t,
                                                                 CDIM, CDIM);

  gemm_qkv<<<dim3(N3 / 128, T_SEQ / 128), 256, 0, stream>>>(
      x_bf, Wqkv_t, bqkv, Qb, Kb, Vt);

  attn_kernel<<<dim3(T_SEQ / 128, NH), 256, 0, stream>>>(Qb, Kb, Vt, Yb);

  gemm_proj<<<dim3(CDIM / 128, T_SEQ / 128), 256, 0, stream>>>(
      Yb, Wproj_t, bproj, out);
}
